// SingleHeadAttention_67843303408050
// MI455X (gfx1250) — matrix operand order repaired and checked by reading, not compiled, not run on hardware
//
#include <hip/hip_runtime.h>

// Problem dimensions (fixed by the reference)
#define LQ 8192
#define LK 8192
#define DM 256
#define DV 64
#define NB 8
#define NSPLIT 16         // column splits for the stats pass (parallelism)
#define KT 128            // key tile for the output pass
#define LDP 136           // LDS row pitch (bf16 elems): 272B -> 16B aligned, bank-shifted
#define LOG2E 1.4426950408889634f

typedef __attribute__((ext_vector_type(16))) __bf16 v16bf;
typedef __attribute__((ext_vector_type(8)))  __bf16 v8bf;
typedef __attribute__((ext_vector_type(8)))  float  v8f;

// Build a 16x32 bf16 A-fragment for one lane: two contiguous 8-element (16B)
// K-chunks (lane L: row = L%16, chunks at K = 16*(L/16) and 16*(L/16)+8).
__device__ __forceinline__ v16bf load_a_bf16(const __bf16* p0, const __bf16* p1) {
  v8bf lo = *(const v8bf*)p0;
  v8bf hi = *(const v8bf*)p1;
  v16bf a;
#pragma unroll
  for (int i = 0; i < 8; ++i) { a[i] = lo[i]; a[i + 8] = hi[i]; }
  return a;
}

// ---------------------------------------------------------------- prep kernels

__global__ __launch_bounds__(256) void cvt_k_kernel(const float* __restrict__ k,
                                                    __bf16* __restrict__ kb) {
  int i = blockIdx.x * blockDim.x + threadIdx.x;          // over n/4 elements
  float4 f = ((const float4*)k)[i];
  __bf16* o = kb + 4 * (size_t)i;
  o[0] = (__bf16)f.x; o[1] = (__bf16)f.y; o[2] = (__bf16)f.z; o[3] = (__bf16)f.w;
}

// v[b][l][d] (f32) -> vt[b][d][l] (bf16), 64x64 LDS tile transpose
__global__ __launch_bounds__(256) void vt_kernel(const float* __restrict__ v,
                                                 __bf16* __restrict__ vt) {
  __shared__ __bf16 t[64][65];
  const int b  = blockIdx.x >> 7;            // 8 batches
  const int l0 = (blockIdx.x & 127) * 64;    // 128 l-tiles
  const int tid = threadIdx.x;
  const int c0 = tid & 63;
  const int r0 = tid >> 6;                   // 0..3
#pragma unroll
  for (int i = 0; i < 16; ++i) {
    int lr = r0 + 4 * i;
    t[lr][c0] = (__bf16)v[((size_t)b * LK + l0 + lr) * DV + c0];
  }
  __syncthreads();
#pragma unroll
  for (int i = 0; i < 16; ++i) {
    int dr = r0 + 4 * i;
    vt[((size_t)b * DV + dr) * LK + l0 + c0] = t[c0][dr];
  }
}

// qw = q @ W^T  (bf16 WMMA, f32 accum), result stored bf16.
// One wave per 16x16 output tile; B column n = W row n (two 8-float chunks per lane, at K = 8*half and 16+8*half).
__global__ __launch_bounds__(256) void qw_kernel(const float* __restrict__ q,
                                                 const float* __restrict__ W,
                                                 __bf16* __restrict__ qw) {
  const int lane = threadIdx.x & 31;
  const int wave = threadIdx.x >> 5;
  const int half = lane >> 4;
  const int l16  = lane & 15;
  const int tile = blockIdx.x * 8 + wave;    // 8192 tiles = (8192/16) x (256/16)
  const int mt = tile >> 4, nt = tile & 15;

  v8f c = {};
#pragma unroll
  for (int kk = 0; kk < 8; ++kk) {
    const float* ar = q + (size_t)(mt * 16 + l16) * DM + kk * 32;
    v16bf a;
#pragma unroll
    for (int i = 0; i < 8; ++i) {
      a[i]     = (__bf16)ar[8 * half + i];
      a[i + 8] = (__bf16)ar[16 + 8 * half + i];
    }
    const float* br = W + (size_t)(nt * 16 + l16) * DM + kk * 32;
    v16bf b;
#pragma unroll
    for (int i = 0; i < 8; ++i) {
      b[i]     = (__bf16)br[8 * half + i];
      b[i + 8] = (__bf16)br[16 + 8 * half + i];
    }
    c = __builtin_amdgcn_wmma_f32_16x16x32_bf16(false, a, false, b, (short)0, c, false, false);
  }
#pragma unroll
  for (int r = 0; r < 8; ++r)
    qw[(size_t)(mt * 16 + r + 8 * half) * DM + nt * 16 + l16] = (__bf16)c[r];
}

// ------------------------------------------------- pass 1: raw scores + stats

// Each wave owns 16 query rows over a 1/NSPLIT column range. Writes raw S to the
// attn output region. Softmax stats kept PER LANE (each lane folds only its own
// columns: 1 max + 2 exp2 + 1 fma per element, NO shuffles in the hot loop);
// the 16-lane log-sum-exp combine happens once at the end. Stats are in log2
// domain so every exponential is a bare v_exp_f32.
__global__ __launch_bounds__(256) void scores_stats_kernel(
    const __bf16* __restrict__ qw, const __bf16* __restrict__ kb,
    float* __restrict__ attn, float* __restrict__ pm, float* __restrict__ pl) {
  const int lane = threadIdx.x & 31;
  const int wave = threadIdx.x >> 5;
  const int half = lane >> 4;
  const int l16  = lane & 15;
  const int m0 = (blockIdx.x * 8 + wave) * 16;
  const int chunk = blockIdx.y;                 // 0..NSPLIT-1
  const int cbase = chunk * (LK / NSPLIT);

  v16bf a[8];                                   // 16 rows x K=256 of qw, resident
#pragma unroll
  for (int kk = 0; kk < 8; ++kk) {
    const __bf16* row = qw + (size_t)(m0 + l16) * DM + kk * 32;
    a[kk] = load_a_bf16(row + 16 * half, row + 16 * half + 8);
  }

  float m_ln[8], l_ln[8];                       // per-lane running stats (log2 domain)
#pragma unroll
  for (int r = 0; r < 8; ++r) { m_ln[r] = -3.0e38f; l_ln[r] = 0.0f; }

  for (int n0 = cbase; n0 < cbase + LK / NSPLIT; n0 += 16) {
    v8f c = {};
#pragma unroll
    for (int kk = 0; kk < 8; ++kk) {
      v16bf b = *(const v16bf*)(kb + (size_t)(n0 + l16) * DM + kk * 32 + 16 * half);
      c = __builtin_amdgcn_wmma_f32_16x16x32_bf16(false, a[kk], false, b, (short)0, c, false, false);
    }
#pragma unroll
    for (int r = 0; r < 8; ++r) {
      float s = c[r];
      attn[(size_t)(m0 + r + 8 * half) * LK + n0 + l16] = s;    // raw scores
      float s2 = s * LOG2E;
      float mo = m_ln[r];
      float mn = fmaxf(mo, s2);
      l_ln[r]  = l_ln[r] * exp2f(mo - mn) + exp2f(s2 - mn);
      m_ln[r]  = mn;
    }
  }

  // cross-lane combine within each 16-lane half (log-sum-exp merge)
#pragma unroll
  for (int r = 0; r < 8; ++r) {
    float m = m_ln[r], l = l_ln[r];
#pragma unroll
    for (int off = 1; off < 16; off <<= 1) {
      float mo = __shfl_xor(m, off, 32);
      float lo = __shfl_xor(l, off, 32);
      float mn = fmaxf(m, mo);
      l = l * exp2f(m - mn) + lo * exp2f(mo - mn);
      m = mn;
    }
    m_ln[r] = m; l_ln[r] = l;
  }
  if (l16 == 0) {
#pragma unroll
    for (int r = 0; r < 8; ++r) {
      pm[(size_t)chunk * LQ + m0 + r + 8 * half] = m_ln[r];
      pl[(size_t)chunk * LQ + m0 + r + 8 * half] = l_ln[r];
    }
  }
}

__global__ __launch_bounds__(256) void merge_stats_kernel(
    const float* __restrict__ pm, const float* __restrict__ pl,
    float* __restrict__ rowm, float* __restrict__ rowl) {
  int r = blockIdx.x * blockDim.x + threadIdx.x;   // 8192 rows
  float m = -3.0e38f;
#pragma unroll
  for (int c = 0; c < NSPLIT; ++c) m = fmaxf(m, pm[(size_t)c * LQ + r]);
  float l = 0.0f;
#pragma unroll
  for (int c = 0; c < NSPLIT; ++c) l += pl[(size_t)c * LQ + r] * exp2f(pm[(size_t)c * LQ + r] - m);
  rowm[r] = m;        // log2-domain row max
  rowl[r] = l;
}

// ----------------------------------- pass 2: normalize attn + out[b] = P @ v[b]

// Workgroup owns 32 query rows. Per 128-key tile: 8 waves cooperatively turn raw
// S into normalized P = exp2(S*log2e - m2)/l (rewritten to attn, staged bf16 in
// LDS), then wave w accumulates out for batch b = w with 32 bf16 WMMAs from the
// L2-resident v^T.
__global__ __launch_bounds__(256) void attn_out_kernel(
    float* __restrict__ attn, const __bf16* __restrict__ vt,
    const float* __restrict__ rowm, const float* __restrict__ rowl,
    float* __restrict__ out) {
  __shared__ __bf16 Pl[32 * LDP];
  const int lane = threadIdx.x & 31;
  const int wave = threadIdx.x >> 5;     // batch index AND P column-block
  const int half = lane >> 4;
  const int l16  = lane & 15;
  const int m0 = blockIdx.x * 32;

  float mrow[2][8], ilrow[2][8];
#pragma unroll
  for (int rh = 0; rh < 2; ++rh)
#pragma unroll
    for (int r = 0; r < 8; ++r) {
      int row = m0 + 16 * rh + r + 8 * half;
      mrow[rh][r]  = rowm[row];
      ilrow[rh][r] = 1.0f / rowl[row];
    }

  v8f acc[2][4];
#pragma unroll
  for (int rh = 0; rh < 2; ++rh)
#pragma unroll
    for (int t = 0; t < 4; ++t) acc[rh][t] = (v8f){};

  for (int k0 = 0; k0 < LK; k0 += KT) {
    // Phase 1: P = exp2(S*log2e - m2) / l ; rewrite attn, stage bf16 in LDS
#pragma unroll
    for (int rh = 0; rh < 2; ++rh) {
#pragma unroll
      for (int r = 0; r < 8; ++r) {
        size_t g = (size_t)(m0 + 16 * rh + r + 8 * half) * LK + k0 + 16 * wave + l16;
        float p = exp2f(fmaf(attn[g], LOG2E, -mrow[rh][r])) * ilrow[rh][r];
        attn[g] = p;
        Pl[(16 * rh + r + 8 * half) * LDP + 16 * wave + l16] = (__bf16)p;
      }
    }
    __syncthreads();
    // Phase 2: acc += P(32x128) @ vT[b](128x64)
#pragma unroll
    for (int kk = 0; kk < 4; ++kk) {
      v16bf aA[2];
#pragma unroll
      for (int rh = 0; rh < 2; ++rh) {
        const __bf16* rp = &Pl[(16 * rh + l16) * LDP + kk * 32];
        aA[rh] = load_a_bf16(rp + 16 * half, rp + 16 * half + 8);
      }
#pragma unroll
      for (int t = 0; t < 4; ++t) {
        v16bf b = *(const v16bf*)(vt + ((size_t)wave * DV + t * 16 + l16) * LK + k0 + kk * 32 + 16 * half);
#pragma unroll
        for (int rh = 0; rh < 2; ++rh)
          acc[rh][t] = __builtin_amdgcn_wmma_f32_16x16x32_bf16(false, aA[rh], false, b, (short)0, acc[rh][t], false, false);
      }
    }
    __syncthreads();
  }

#pragma unroll
  for (int rh = 0; rh < 2; ++rh)
#pragma unroll
    for (int t = 0; t < 4; ++t)
#pragma unroll
      for (int r = 0; r < 8; ++r)
        out[((size_t)wave * LQ + m0 + 16 * rh + r + 8 * half) * DV + t * 16 + l16] = acc[rh][t][r];
}

// ---------------------------------------------------------------------- launch

extern "C" void kernel_launch(void* const* d_in, const int* in_sizes, int n_in,
                              void* d_out, int out_size, void* d_ws, size_t ws_size,
                              hipStream_t stream) {
  (void)in_sizes; (void)n_in; (void)out_size; (void)ws_size;
  const float* q = (const float*)d_in[0];
  const float* k = (const float*)d_in[1];
  const float* v = (const float*)d_in[2];
  const float* W = (const float*)d_in[3];

  float* out  = (float*)d_out;                        // [8, 8192, 64]
  float* attn = (float*)d_out + (size_t)NB * LQ * DV; // [8192, 8192]

  // Workspace layout (~17.1 MB total)
  char* w = (char*)d_ws;
  __bf16* qw   = (__bf16*)(w);                          // 4 MB
  __bf16* kb   = (__bf16*)(w + (4u << 20));             // 4 MB
  __bf16* vt   = (__bf16*)(w + (8u << 20));             // 8 MB
  float*  pm   = (float*)(w + (16u << 20));             // 512 KB (NSPLIT x 8192)
  float*  pl   = (float*)(w + (16u << 20) + (512u << 10));
  float*  rowm = (float*)(w + (17u << 20));             // 32 KB
  float*  rowl = (float*)(w + (17u << 20) + (32u << 10));

  cvt_k_kernel<<<(LK * DM / 4) / 256, 256, 0, stream>>>(k, kb);
  vt_kernel<<<NB * (LK / 64), 256, 0, stream>>>(v, vt);
  qw_kernel<<<(LQ / 16) * (DM / 16) / 8, 256, 0, stream>>>(q, W, qw);
  scores_stats_kernel<<<dim3(LQ / 16 / 8, NSPLIT), 256, 0, stream>>>(qw, kb, attn, pm, pl);
  merge_stats_kernel<<<LQ / 256, 256, 0, stream>>>(pm, pl, rowm, rowl);
  attn_out_kernel<<<LQ / 32, 256, 0, stream>>>(attn, vt, rowm, rowl, out);
}
